// LIFLayer_90108413870273
// MI455X (gfx1250) — matrix operand order repaired and checked by reading, not compiled, not run on hardware
//
#include <hip/hip_runtime.h>
#include <math.h>

#define B_   128
#define T_   256
#define DI   512
#define DO   512
#define BT   (B_*T_)

typedef __attribute__((ext_vector_type(16))) __bf16 v16bf;
typedef __attribute__((ext_vector_type(8)))  __bf16 v8bf;
typedef __attribute__((ext_vector_type(8)))  float  v8f;
typedef __attribute__((ext_vector_type(4)))  float  v4f;

#define MBLK 128
#define NBLK 64
#define KS   32
#define APAD 40   // LDS row stride (elems): 80B = 20 dwords -> conflict-free b128 loads
#define WPAD 40

// ---------------------------------------------------------------------------
// Kernel 1: syn[bt, o] = sum_i spikes[bt, i] * W[o, i]
// Split-bf16 WMMA: W = Whi + Wlo (bf16 each), A (0/1 spikes) exact in bf16.
// Output written into the 'traces' half of d_out (consumed in-place by scan).
// ---------------------------------------------------------------------------
__global__ __launch_bounds__(256)
void lif_gemm(const float* __restrict__ X, const float* __restrict__ W,
              float* __restrict__ synOut) {
  __shared__ __bf16 As[MBLK][APAD];
  __shared__ __bf16 Bh[NBLK][WPAD];
  __shared__ __bf16 Bl[NBLK][WPAD];

  const int tid  = threadIdx.x;
  const int lane = tid & 31;
  const int wid  = tid >> 5;
  const int wm   = wid & 3;            // wave M index (0..3)  -> 32-row slab
  const int wn   = wid >> 2;           // wave N index (0..1)  -> 32-col slab
  const int m0   = blockIdx.x * MBLK;
  const int n0   = blockIdx.y * NBLK;

  v8f c[2][2] = {};

  for (int k0 = 0; k0 < DI; k0 += KS) {
    // ---- stage A tile (128x32 f32 -> bf16), spikes are exactly 0/1
#pragma unroll
    for (int it = 0; it < 4; ++it) {
      int s  = tid + it * 256;         // 0..1023
      int r  = s >> 3;                 // 0..127
      int cc = (s & 7) * 4;            // 0,4,..,28
      v4f x = *(const v4f*)(X + (size_t)(m0 + r) * DI + k0 + cc);
      As[r][cc + 0] = (__bf16)x.x;  As[r][cc + 1] = (__bf16)x.y;
      As[r][cc + 2] = (__bf16)x.z;  As[r][cc + 3] = (__bf16)x.w;
    }
    // ---- stage W tile (64x32 f32 -> bf16 hi + lo)
#pragma unroll
    for (int it = 0; it < 2; ++it) {
      int s  = tid + it * 256;         // 0..511
      int r  = s >> 3;                 // 0..63
      int cc = (s & 7) * 4;
      v4f w = *(const v4f*)(W + (size_t)(n0 + r) * DI + k0 + cc);
#pragma unroll
      for (int j = 0; j < 4; ++j) {
        float wf = (j == 0) ? w.x : (j == 1) ? w.y : (j == 2) ? w.z : w.w;
        __bf16 h = (__bf16)wf;
        __bf16 l = (__bf16)(wf - (float)h);
        Bh[r][cc + j] = h;
        Bl[r][cc + j] = l;
      }
    }
    __syncthreads();

    // ---- fragment loads per ISA 7.12.2 16-bit layouts
    const int arow0 = wm * 32 + (lane & 15);
    const int akoff = (lane >> 4) ? 8 : 0;    // K 0-7/16-23 vs 8-15/24-31
    const int bcol0 = wn * 32 + (lane & 15);
    const int bkoff = (lane >> 4) ? 8 : 0;    // K 0-7/16-23 vs 8-15/24-31

    union F { v16bf v; v8bf h[2]; };

#pragma unroll
    for (int mi = 0; mi < 2; ++mi) {
      F a;
      const __bf16* ap = &As[arow0 + mi * 16][0];
      a.h[0] = *(const v8bf*)(ap + akoff);
      a.h[1] = *(const v8bf*)(ap + akoff + 16);
#pragma unroll
      for (int ni = 0; ni < 2; ++ni) {
        F bh, bl;
        const __bf16* bph = &Bh[bcol0 + ni * 16][0];
        const __bf16* bpl = &Bl[bcol0 + ni * 16][0];
        bh.h[0] = *(const v8bf*)(bph + bkoff);
        bh.h[1] = *(const v8bf*)(bph + bkoff + 16);
        bl.h[0] = *(const v8bf*)(bpl + bkoff);
        bl.h[1] = *(const v8bf*)(bpl + bkoff + 16);
        c[mi][ni] = __builtin_amdgcn_wmma_f32_16x16x32_bf16(
            false, a.v, false, bh.v, (short)0, c[mi][ni], false, false);
        c[mi][ni] = __builtin_amdgcn_wmma_f32_16x16x32_bf16(
            false, a.v, false, bl.v, (short)0, c[mi][ni], false, false);
      }
    }
    __syncthreads();
  }

  // ---- store C: VGPR j -> row (j + 8*(lane>=16)), col (lane&15)
  const int rsel = (lane >> 4) * 8;
  const int csel = lane & 15;
#pragma unroll
  for (int mi = 0; mi < 2; ++mi)
#pragma unroll
    for (int ni = 0; ni < 2; ++ni) {
      const int gm = m0 + wm * 32 + mi * 16 + rsel;
      const int gn = n0 + wn * 32 + ni * 16 + csel;
#pragma unroll
      for (int j = 0; j < 8; ++j)
        synOut[(size_t)(gm + j) * DO + gn] = c[mi][ni][j];  // stays L2-resident
    }
}

// ---------------------------------------------------------------------------
// Kernel 2: sequential LIF scan per (b, o); reads syn from the traces buffer
// and overwrites it in place with the membrane trace. Fully coalesced over o.
// ---------------------------------------------------------------------------
__global__ __launch_bounds__(256)
void lif_scan(const float* __restrict__ decay_raw, const float* __restrict__ thr,
              float* __restrict__ spikes, float* __restrict__ traces) {
  const int gid = blockIdx.x * 256 + threadIdx.x;   // 0 .. B*O-1
  const int o = gid & (DO - 1);
  const int b = gid >> 9;                           // DO == 512

  const float decay = 1.0f / (1.0f + __expf(-decay_raw[o]));
  const float th    = fabsf(thr[o]);
  float mem = 0.0f;

  float* tr = traces + (size_t)b * T_ * DO + o;
  float* sp = spikes + (size_t)b * T_ * DO + o;

#pragma unroll 4
  for (int t = 0; t < T_; ++t) {
    const float syn = tr[(size_t)t * DO];           // L2 hit (written by GEMM)
    mem = __builtin_fmaf(decay, mem, syn);
    const float s = (mem >= th) ? 1.0f : 0.0f;
    __builtin_nontemporal_store(s,   &sp[(size_t)t * DO]);
    __builtin_nontemporal_store(mem, &tr[(size_t)t * DO]);
    mem = (s > 0.0f) ? 0.0f : mem;                  // reset on spike
  }
}

extern "C" void kernel_launch(void* const* d_in, const int* in_sizes, int n_in,
                              void* d_out, int out_size, void* d_ws, size_t ws_size,
                              hipStream_t stream) {
  (void)in_sizes; (void)n_in; (void)out_size; (void)d_ws; (void)ws_size;
  const float* X         = (const float*)d_in[0];   // (B, T, DI) spikes
  const float* W         = (const float*)d_in[1];   // (DO, DI)
  const float* decay_raw = (const float*)d_in[2];   // (DO,)
  const float* thr       = (const float*)d_in[3];   // (DO,)

  float* spikes = (float*)d_out;                    // (B, T, DO)
  float* traces = spikes + (size_t)B_ * T_ * DO;    // (B, T, DO), also syn buffer

  dim3 g1(BT / MBLK, DO / NBLK);                    // 256 x 8 = 2048 blocks
  lif_gemm<<<g1, 256, 0, stream>>>(X, W, traces);
  lif_scan<<<(B_ * DO) / 256, 256, 0, stream>>>(decay_raw, thr, spikes, traces);
}
